// LSTMnLayers_8392366096810
// MI455X (gfx1250) — matrix operand order repaired and checked by reading, not compiled, not run on hardware
//
#include <hip/hip_runtime.h>
#include <hip/hip_bf16.h>

typedef __attribute__((ext_vector_type(16))) _Float16 v16h;
typedef __attribute__((ext_vector_type(8)))  _Float16 v8h;
typedef __attribute__((ext_vector_type(8)))  float    v8f;

#define BATCH 1024
#define TSTEPS 512
#define EDIM 32
#define HDIM 128
#define LDX0 160   // E + H
#define LDX1 256   // 2H
#define KS0 5      // 160 / 32
#define KS1 8      // 256 / 32
#define FRAG_HALFS 512          // 32 lanes * 16 halfs
#define L0_FRAGS (4*8*KS0)      // 160 frags (160 KB f16)
#define L1_FRAGS (4*8*KS1)      // 256 frags (256 KB f16)
#define PACK_TOTAL ((L0_FRAGS + L1_FRAGS) * FRAG_HALFS) // 212992 halfs

struct WPtrs  { const float* w[8]; };   // Wf0,Wi0,Wo0,Wc0,Wf1,Wi1,Wo1,Wc1
struct BPtrs  { const float* b[8]; };   // bf0,bi0,bo0,bc0,bf1,bi1,bo1,bc1

// ---------------------------------------------------------------------------
// Pack fp32 weights into f16 WMMA B-fragment order (32x16 KxN per fragment):
//   lane l: column n = l & 15 ; K half-group = (l>>4) ? 16..31 : 0..15
//   each lane holds 16 contiguous K values (32 B) -> two coalesced b128 loads.
// ---------------------------------------------------------------------------
__global__ __launch_bounds__(256) void lstm_pack_weights(WPtrs wp, _Float16* __restrict__ outw)
{
    int id = blockIdx.x * 256 + threadIdx.x;
    if (id < L0_FRAGS * FRAG_HALFS) {
        int f   = id / FRAG_HALFS;
        int rem = id % FRAG_HALFS;
        int l = rem >> 4, j = rem & 15;
        int gn = f / KS0, ks = f % KS0;
        int g = gn >> 3, n = gn & 7;
        int col = n * 16 + (l & 15);
        int K   = ks * 32 + ((l >> 4) & 1) * 16 + j;       // K in [0,160)
        outw[id] = (_Float16)wp.w[g][K * HDIM + col];
    } else if (id < PACK_TOTAL) {
        int id1 = id - L0_FRAGS * FRAG_HALFS;
        int f   = id1 / FRAG_HALFS;
        int rem = id1 % FRAG_HALFS;
        int l = rem >> 4, j = rem & 15;
        int gn = f / KS1, ks = f % KS1;
        int g = gn >> 3, n = gn & 7;
        int col = n * 16 + (l & 15);
        int K   = ks * 32 + ((l >> 4) & 1) * 16 + j;       // K in [0,256)
        outw[id] = (_Float16)wp.w[4 + g][K * HDIM + col];
    }
}

// gfx1250 hardware transcendental tanh (TRANS op, co-executes with WMMA)
__device__ __forceinline__ float fast_tanh(float x) {
    float r;
    asm volatile("v_tanh_f32 %0, %1" : "=v"(r) : "v"(x));
    return r;
}
__device__ __forceinline__ float sigf(float x) {
    return __builtin_fmaf(0.5f, fast_tanh(0.5f * x), 0.5f);
}

// ---------------------------------------------------------------------------
// Main persistent LSTM kernel. One block = 16 batch rows, 8 waves.
// Wave w owns output columns [16w, 16w+16) of H for all 4 gates, both layers.
// Layer-1 weights staged once into LDS (256 KB of the 320 KB WGP LDS);
// layer-0 weights stream from L2 each step (laundered offset defeats LICM).
// ---------------------------------------------------------------------------
__global__ __launch_bounds__(256) void lstm_main(
    const int*      __restrict__ data,   // (B, T)
    const float*    __restrict__ emb,    // (VOCAB, E)
    const _Float16* __restrict__ wpack,  // packed B fragments in d_ws
    BPtrs bp,
    const float*    __restrict__ Wout,   // (H, 1)
    const float*    __restrict__ bout,   // (1,)
    float*          __restrict__ out)    // (B, 1)
{
    __shared__ alignas(32) _Float16 wl1[L1_FRAGS * FRAG_HALFS]; // 256 KB layer-1 frags
    __shared__ alignas(32) _Float16 xh0[16 * LDX0];             // [xt | h0] f16
    __shared__ alignas(32) _Float16 xh1[16 * LDX1];             // [h0 | h1] f16
    __shared__ int s_cur[16];
    __shared__ int s_last[16];

    const int tid  = threadIdx.x;
    const int lane = tid & 31;
    const int wave = tid >> 5;           // ntile 0..7
    const int b0   = blockIdx.x * 16;
    const int cn   = lane & 15;          // column within tile (B/C/D layout)
    const int hi   = (lane >> 4) & 1;    // lane-high flag
    const int rowbase = hi * 8;          // C/D row base for this half-wave

    // ---- one-time stage of layer-1 fragments into LDS --------------------
    {
        const v8h* src = (const v8h*)(wpack + (long)L0_FRAGS * FRAG_HALFS);
        v8h*       dst = (v8h*)wl1;
        for (int i = tid; i < (L1_FRAGS * FRAG_HALFS) / 8; i += 256) dst[i] = src[i];
    }
    for (int i = tid; i < 16 * LDX0; i += 256) xh0[i] = (_Float16)0.0f;
    for (int i = tid; i < 16 * LDX1; i += 256) xh1[i] = (_Float16)0.0f;
    if (tid < 16) s_last[tid] = data[(b0 + tid) * TSTEPS + (TSTEPS - 1)];
    __syncthreads();

    v8f c0 = {}, h0 = {}, c1 = {}, h1 = {};

    // per-lane gate biases (column-constant across rows)
    const int gc = wave * 16 + cn;
    float bv0[4], bv1[4];
#pragma unroll
    for (int g = 0; g < 4; ++g) { bv0[g] = bp.b[g][gc]; bv1[g] = bp.b[4 + g][gc]; }

    // A-fragment loader: contiguous k = 16*hi + s per lane (matches B packing), two 16B loads
    auto loadA = [&](const _Float16* base, int ld, int ks) -> v16h {
        const _Float16* p = base + (lane & 15) * ld + ks * 32 + hi * 16;
        v8h lo = *(const v8h*)(p);
        v8h hc = *(const v8h*)(p + 8);
        return __builtin_shufflevector(lo, hc, 0,1,2,3,4,5,6,7,8,9,10,11,12,13,14,15);
    };

    for (int t = 0; t < TSTEPS; ++t) {
        // launder a zero offset each iteration: keeps weight loads in-loop
        // (no LICM -> no 400-VGPR hoist -> no scratch spills)
        int zoff = 0;
        asm volatile("" : "+s"(zoff));

        if (tid < 16) s_cur[tid] = data[(b0 + tid) * TSTEPS + t];
        __syncthreads();

        // gather current-step embeddings into xh0 x-part (16 x 32 f16)
        for (int i = tid; i < 16 * EDIM; i += 256) {
            int m = i >> 5, e = i & 31;
            xh0[m * LDX0 + e] = (_Float16)emb[(long)s_cur[m] * EDIM + e];
        }
        bool msk[8];
#pragma unroll
        for (int r = 0; r < 8; ++r)
            msk[r] = (s_cur[r + rowbase] != s_last[r + rowbase]);
        __syncthreads();

        // ------------------ layer 0: xh0(16x160) @ W0(160x512), B from L2 --
        v8f acc0 = {}, acc1 = {}, acc2 = {}, acc3 = {};
#pragma unroll
        for (int ks = 0; ks < KS0; ++ks) {
            v16h a = loadA(xh0, LDX0, ks);
            const _Float16* pb = wpack + zoff + lane * 16;
            v16h bf = *(const v16h*)(pb + ((0*8 + wave) * KS0 + ks) * FRAG_HALFS);
            v16h bi = *(const v16h*)(pb + ((1*8 + wave) * KS0 + ks) * FRAG_HALFS);
            v16h bo = *(const v16h*)(pb + ((2*8 + wave) * KS0 + ks) * FRAG_HALFS);
            v16h bc = *(const v16h*)(pb + ((3*8 + wave) * KS0 + ks) * FRAG_HALFS);
            acc0 = __builtin_amdgcn_wmma_f32_16x16x32_f16(false, a, false, bf, (short)0, acc0, false, false);
            acc1 = __builtin_amdgcn_wmma_f32_16x16x32_f16(false, a, false, bi, (short)0, acc1, false, false);
            acc2 = __builtin_amdgcn_wmma_f32_16x16x32_f16(false, a, false, bo, (short)0, acc2, false, false);
            acc3 = __builtin_amdgcn_wmma_f32_16x16x32_f16(false, a, false, bc, (short)0, acc3, false, false);
        }
        __syncthreads();   // all waves done reading xh0 before h0 writeback
#pragma unroll
        for (int r = 0; r < 8; ++r) {
            float f  = sigf(acc0[r] + bv0[0]);
            float ii = sigf(acc1[r] + bv0[1]);
            float o  = sigf(acc2[r] + bv0[2]);
            float g  = fast_tanh(acc3[r] + bv0[3]);
            float cnw = c0[r] * f + ii * g;
            float hnw = o * fast_tanh(cnw);
            c0[r] = msk[r] ? cnw : c0[r];
            h0[r] = msk[r] ? hnw : h0[r];
            int row = r + rowbase;
            _Float16 hh = (_Float16)h0[r];
            xh0[row * LDX0 + EDIM + gc] = hh;   // next step's layer0 input
            xh1[row * LDX1 + gc]        = hh;   // this step's layer1 input
        }
        __syncthreads();

        // ------------------ layer 1: xh1(16x256) @ W1(256x512), B from LDS -
        acc0 = v8f{}; acc1 = v8f{}; acc2 = v8f{}; acc3 = v8f{};
#pragma unroll
        for (int ks = 0; ks < KS1; ++ks) {
            v16h a = loadA(xh1, LDX1, ks);
            const _Float16* pb = wl1 + zoff + lane * 16;
            v16h bf = *(const v16h*)(pb + ((0*8 + wave) * KS1 + ks) * FRAG_HALFS);
            v16h bi = *(const v16h*)(pb + ((1*8 + wave) * KS1 + ks) * FRAG_HALFS);
            v16h bo = *(const v16h*)(pb + ((2*8 + wave) * KS1 + ks) * FRAG_HALFS);
            v16h bc = *(const v16h*)(pb + ((3*8 + wave) * KS1 + ks) * FRAG_HALFS);
            acc0 = __builtin_amdgcn_wmma_f32_16x16x32_f16(false, a, false, bf, (short)0, acc0, false, false);
            acc1 = __builtin_amdgcn_wmma_f32_16x16x32_f16(false, a, false, bi, (short)0, acc1, false, false);
            acc2 = __builtin_amdgcn_wmma_f32_16x16x32_f16(false, a, false, bo, (short)0, acc2, false, false);
            acc3 = __builtin_amdgcn_wmma_f32_16x16x32_f16(false, a, false, bc, (short)0, acc3, false, false);
        }
        __syncthreads();   // all waves done reading xh1 before h1 writeback
#pragma unroll
        for (int r = 0; r < 8; ++r) {
            float f  = sigf(acc0[r] + bv1[0]);
            float ii = sigf(acc1[r] + bv1[1]);
            float o  = sigf(acc2[r] + bv1[2]);
            float g  = fast_tanh(acc3[r] + bv1[3]);
            float cnw = c1[r] * f + ii * g;
            float hnw = o * fast_tanh(cnw);
            c1[r] = msk[r] ? cnw : c1[r];
            h1[r] = msk[r] ? hnw : h1[r];
            int row = r + rowbase;
            xh1[row * LDX1 + HDIM + gc] = (_Float16)h1[r];
        }
        __syncthreads();
    }

    // ---- output head: sigmoid(h1 @ Wout + bout) --------------------------
    if (tid < 16) {
        float s = bout[0];
        for (int k = 0; k < HDIM; ++k)
            s += (float)xh1[tid * LDX1 + HDIM + k] * Wout[k];
        out[b0 + tid] = sigf(s);
    }
}

extern "C" void kernel_launch(void* const* d_in, const int* in_sizes, int n_in,
                              void* d_out, int out_size, void* d_ws, size_t ws_size,
                              hipStream_t stream)
{
    (void)in_sizes; (void)n_in; (void)out_size; (void)ws_size;
    const int*   data = (const int*)  d_in[0];
    const float* emb  = (const float*)d_in[1];

    WPtrs wp; BPtrs bp;
    for (int g = 0; g < 4; ++g) {
        wp.w[g]     = (const float*)d_in[2  + 2 * g];   // Wf0,Wi0,Wo0,Wc0
        bp.b[g]     = (const float*)d_in[3  + 2 * g];   // bf0,bi0,bo0,bc0
        wp.w[4 + g] = (const float*)d_in[10 + 2 * g];   // Wf1,Wi1,Wo1,Wc1
        bp.b[4 + g] = (const float*)d_in[11 + 2 * g];   // bf1,bi1,bo1,bc1
    }
    const float* Wout = (const float*)d_in[18];
    const float* bout = (const float*)d_in[19];

    _Float16* wpack = (_Float16*)d_ws;   // 416 KB packed fragments

    lstm_pack_weights<<<PACK_TOTAL / 256, 256, 0, stream>>>(wp, wpack);
    lstm_main<<<BATCH / 16, 256, 0, stream>>>(data, emb, wpack, bp, Wout, bout, (float*)d_out);
}
